// KA_attention_crossinf_2layer_20761871909462
// MI455X (gfx1250) — matrix operand order repaired and checked by reading, not compiled, not run on hardware
//
#include <hip/hip_runtime.h>
#include <hip/hip_bf16.h>
#include <math.h>

// ---------------------------------------------------------------------------
// Problem constants (from reference): B=4 H=8 P=196 D=64 NF=8
// ---------------------------------------------------------------------------
#define Bc 4
#define Hc 8
#define Pc 196
#define Dc 64
#define NFc 8
#define BH (Bc*Hc)          // 32
#define ROWS (BH*Pc)        // 6272
#define PP (Pc*Pc)          // 38416

typedef __attribute__((ext_vector_type(16))) __bf16 v16bf;
typedef __attribute__((ext_vector_type(8)))  float  v8f;

__device__ __forceinline__ float silu_f(float x) {
    return x / (1.0f + __expf(-x));
}

// Convert 4 floats to 4 bf16 slots of a v16bf fragment, optional SiLU.
template<bool SILU>
__device__ __forceinline__ void put4(v16bf& f, int h, float4 v) {
    if (SILU) { v.x = silu_f(v.x); v.y = silu_f(v.y); v.z = silu_f(v.z); v.w = silu_f(v.w); }
    f[h+0] = (__bf16)v.x; f[h+1] = (__bf16)v.y; f[h+2] = (__bf16)v.z; f[h+3] = (__bf16)v.w;
}

// ---------------------------------------------------------------------------
// WMMA GEMM:  C[M,N] = act(A[M,64]) @ W[N,64]^T (+ bias[N])
// One 16x16 output tile per wave; K=64 as two K=32 bf16 WMMA steps.
// Grid: (Mtiles/4, Ntiles), block = 128 threads (4 waves).
// Fragment layouts per CDNA5 ISA 7.12.2:
//   A (16-bit 16x32): lane m=l%16; half h -> K = h + (h>=8?8:0) + (l>=16?8:0)
//   B (16-bit 32x16): lane n=l%16; half h -> K = h + (h>=8?8:0) + (l>=16?8:0)
//   C/D (f32 16x16):  vgpr r -> M = r + (l>=16?8:0); N = l%16
// ---------------------------------------------------------------------------
template<bool SILU>
__global__ void wmma_gemm_k64(const float* __restrict__ A,
                              const float* __restrict__ W,
                              const float* __restrict__ bias,
                              float* __restrict__ C,
                              int M, int N)
{
    const int wave = threadIdx.x >> 5;
    const int lane = threadIdx.x & 31;
    const int hi16 = (lane >= 16);
    const int mBase = (blockIdx.x * 4 + wave) * 16;
    const int nBase = blockIdx.y * 16;

    const int m = mBase + (lane & 15);
    const int n = nBase + (lane & 15);

    v8f acc = {};
    #pragma unroll
    for (int kb = 0; kb < 64; kb += 32) {
        // ---- A fragment: two contiguous 8-float runs ----
        const float* ap = A + (size_t)m * 64 + kb + (hi16 ? 8 : 0);
        v16bf fa;
        put4<SILU>(fa, 0,  *(const float4*)(ap +  0));
        put4<SILU>(fa, 4,  *(const float4*)(ap +  4));
        put4<SILU>(fa, 8,  *(const float4*)(ap + 16));
        put4<SILU>(fa, 12, *(const float4*)(ap + 20));
        // ---- B fragment: two contiguous 8-float runs of W row n ----
        const float* wp = W + (size_t)n * 64 + kb + (hi16 ? 8 : 0);
        v16bf fb;
        put4<false>(fb, 0,  *(const float4*)(wp +  0));
        put4<false>(fb, 4,  *(const float4*)(wp +  4));
        put4<false>(fb, 8,  *(const float4*)(wp + 16));
        put4<false>(fb, 12, *(const float4*)(wp + 20));
        acc = __builtin_amdgcn_wmma_f32_16x16x32_bf16(
                  false, fa, false, fb, (short)0, acc, false, false);
    }

    const int nc = nBase + (lane & 15);
    const float bv = bias ? bias[nc] : 0.0f;
    #pragma unroll
    for (int r = 0; r < 8; ++r) {
        const int mr = mBase + r + (hi16 ? 8 : 0);
        C[(size_t)mr * N + nc] = acc[r] + bv;
    }
}

// ---------------------------------------------------------------------------
// Stage 2: per (bh, s2, p2) compute a = sum_d sigmoid( f_sin*scale_sp +
//          base*scale_base ) with the torch-style raw-reshape index scramble.
// x5   : [6272,5,64] f32  (q5 or k5)
// basex: [6272,5,64] f32  (silu(x5) @ base_weight^T, raw row order)
// coefx: [64,8,5]  gridv: [5,8]  scales: [1,H,5,P,D]
// ax   : [BH*5*196]  flat  bh*980 + s2*196 + p2
// grid = BH*5 blocks, block = 224 threads (196 active)
// ---------------------------------------------------------------------------
__global__ void stage2_activations(const float* __restrict__ x5,
                                   const float* __restrict__ basex,
                                   const float* __restrict__ coefx,
                                   const float* __restrict__ gridv,
                                   const float* __restrict__ scale_sp,
                                   const float* __restrict__ scale_base,
                                   float* __restrict__ ax)
{
    const int bh = blockIdx.x / 5;
    const int s2 = blockIdx.x % 5;
    const int h  = bh % Hc;
    const int p2 = threadIdx.x;
    if (p2 >= Pc) return;

    // Gather descriptors for the 8 frequency terms (raw reshape scramble).
    const float* rowp[NFc];
    float gs[NFc];
    int   coff[NFc];
    #pragma unroll
    for (int n2 = 0; n2 < NFc; ++n2) {
        const int j = s2 * (NFc * Pc) + n2 * Pc + p2;   // s2*1568 + n2*196 + p2
        const int p = j / 40;
        const int s = (j % 40) / 8;
        const int n = j % 8;
        rowp[n2] = x5 + ((size_t)(bh * Pc + p) * 5 + s) * Dc;
        gs[n2]   = gridv[s * NFc + n];
        coff[n2] = n * 5 + s;
    }
    // Base-path scramble.
    const int j2 = s2 * Pc + p2;
    const int pb = j2 / 5, sb = j2 % 5;
    const float* bp = basex + ((size_t)(bh * Pc + pb) * 5 + sb) * Dc;
    const size_t sc = ((size_t)(h * 5 + s2) * Pc + p2) * Dc;

    float acc = 0.0f;
    for (int d = 0; d < Dc; ++d) {
        float fs = 0.0f;
        #pragma unroll
        for (int n2 = 0; n2 < NFc; ++n2)
            fs += __sinf(gs[n2] * rowp[n2][d]) * coefx[d * 40 + coff[n2]];
        const float f = fs * scale_sp[sc + d] + bp[d] * scale_base[sc + d];
        acc += 1.0f / (1.0f + __expf(-f));
    }
    ax[(size_t)blockIdx.x * Pc + p2] = acc;
}

// ---------------------------------------------------------------------------
// Stage 3: outer sin expansion + weighted sum over s + row softmax.
// One block per (bh, p) row; 224 threads, p' = threadIdx.x.
//  f[p'] = sum_s coef_outer[s,pp] * sum_n sin(grid[s,n]*(aq[bh,s,p]+ak[bh,s,p'])) * cqk[pp,n,s]
// coef_qk: [PP,8,5] contiguous 40 floats per pp. Then softmax over p'.
// ---------------------------------------------------------------------------
__global__ void stage3_outer_softmax(const float* __restrict__ aq,
                                     const float* __restrict__ ak,
                                     const float* __restrict__ gridv,
                                     const float* __restrict__ coef_qk,
                                     const float* __restrict__ couter,
                                     float* __restrict__ out)
{
    __shared__ float red[224];
    const int bh = blockIdx.x / Pc;
    const int p  = blockIdx.x % Pc;
    const int t  = threadIdx.x;
    const bool valid = (t < Pc);

    float aqs[5];
    #pragma unroll
    for (int s = 0; s < 5; ++s)
        aqs[s] = aq[(size_t)bh * 980 + s * Pc + p];

    float v = 0.0f;
    if (valid) {
        const int pp = p * Pc + t;
        const float* cp = coef_qk + (size_t)pp * 40;
        __builtin_prefetch(cp, 0, 0);
        #pragma unroll
        for (int s = 0; s < 5; ++s) {
            const float qk = aqs[s] + ak[(size_t)bh * 980 + s * Pc + t];
            float inner = 0.0f;
            #pragma unroll
            for (int n = 0; n < NFc; ++n)
                inner += __sinf(gridv[s * NFc + n] * qk) * cp[n * 5 + s];
            v += inner * couter[(size_t)s * PP + pp];
        }
    }

    // ---- block softmax over 196 valid lanes ----
    red[t] = valid ? v : -3.4e38f;
    __syncthreads();
    for (int s = 128; s > 0; s >>= 1) {
        if (t < s && t + s < 224) red[t] = fmaxf(red[t], red[t + s]);
        __syncthreads();
    }
    const float mx = red[0];
    __syncthreads();
    const float e = valid ? __expf(v - mx) : 0.0f;
    red[t] = e;
    __syncthreads();
    for (int s = 128; s > 0; s >>= 1) {
        if (t < s && t + s < 224) red[t] += red[t + s];
        __syncthreads();
    }
    const float inv = 1.0f / red[0];
    if (valid) out[(size_t)blockIdx.x * Pc + t] = e * inv;
}

// ---------------------------------------------------------------------------
extern "C" void kernel_launch(void* const* d_in, const int* in_sizes, int n_in,
                              void* d_out, int out_size, void* d_ws, size_t ws_size,
                              hipStream_t stream)
{
    const float* q          = (const float*)d_in[0];
    const float* k          = (const float*)d_in[1];
    /* d_in[2] = scale: accepted but unused (as in reference) */
    const float* gridv      = (const float*)d_in[3];
    const float* grid_outer = (const float*)d_in[4];
    const float* base_w     = (const float*)d_in[5];
    const float* coef_q     = (const float*)d_in[6];
    const float* coef_k     = (const float*)d_in[7];
    const float* coef_qk    = (const float*)d_in[8];
    const float* scale_base = (const float*)d_in[9];
    const float* scale_sp   = (const float*)d_in[10];
    const float* couter     = (const float*)d_in[11];
    const float* lin_q_w    = (const float*)d_in[12];
    const float* lin_q_b    = (const float*)d_in[13];
    const float* lin_k_w    = (const float*)d_in[14];
    const float* lin_k_b    = (const float*)d_in[15];
    float* out = (float*)d_out;

    // Workspace carve-up (floats).
    float* ws  = (float*)d_ws;
    float* q5  = ws;                      // [6272,320]
    float* k5  = q5 + (size_t)ROWS * 320;
    float* bq  = k5 + (size_t)ROWS * 320; // [31360,64]
    float* bk  = bq + (size_t)ROWS * 5 * 64;
    float* aq  = bk + (size_t)ROWS * 5 * 64; // [32*5*196]
    float* ak  = aq + (size_t)BH * 5 * Pc;

    // --- Stage 1: linear layers, bf16 WMMA, f32 accumulate ---
    // q5 = q @ lin_q_w^T + b : M=6272 (392 tiles -> 98 blocks x 4 waves), N=320 (20 tiles)
    wmma_gemm_k64<false><<<dim3(98, 20), 128, 0, stream>>>(q, lin_q_w, lin_q_b, q5, ROWS, 320);
    wmma_gemm_k64<false><<<dim3(98, 20), 128, 0, stream>>>(k, lin_k_w, lin_k_b, k5, ROWS, 320);
    // base = silu(x5) @ base_weight^T : M=31360 (1960 tiles -> 490 blocks), N=64 (4 tiles)
    wmma_gemm_k64<true ><<<dim3(490, 4), 128, 0, stream>>>(q5, base_w, nullptr, bq, ROWS * 5, 64);
    wmma_gemm_k64<true ><<<dim3(490, 4), 128, 0, stream>>>(k5, base_w, nullptr, bk, ROWS * 5, 64);

    // --- Stage 2: sin-basis + base mix + sigmoid row-sums ---
    stage2_activations<<<BH * 5, 224, 0, stream>>>(q5, bq, coef_q, gridv, scale_sp, scale_base, aq);
    stage2_activations<<<BH * 5, 224, 0, stream>>>(k5, bk, coef_k, gridv, scale_sp, scale_base, ak);

    // --- Stage 3: outer expansion + softmax (one block per output row) ---
    stage3_outer_softmax<<<BH * Pc, 224, 0, stream>>>(aq, ak, grid_outer, coef_qk, couter, out);
}
